// VPRNetwork_70978629534004
// MI455X (gfx1250) — matrix operand order repaired and checked by reading, not compiled, not run on hardware
//
#include <hip/hip_runtime.h>
#include <hip/hip_bf16.h>

typedef __attribute__((ext_vector_type(16))) _Float16 v16h;
typedef __attribute__((ext_vector_type(8)))  float    v8f;
typedef __attribute__((ext_vector_type(4)))  float    f32x4;

#define BM 128
#define BN 64
#define BK 32

// ---------------------------------------------------------------------------
// Async global -> LDS copy of 16 bytes (per lane), tracked by ASYNCcnt.
// lds_off = wave-relative LDS byte offset (low 32 bits of generic pointer).
// ---------------------------------------------------------------------------
__device__ __forceinline__ void async_copy_b128(const void* lds_ptr, const void* gptr)
{
    const uint32_t lds_off = (uint32_t)(uintptr_t)lds_ptr;
    const uint64_t gaddr   = (uint64_t)(uintptr_t)gptr;
    asm volatile("global_load_async_to_lds_b128 %0, %1, off"
                 :: "v"(lds_off), "v"(gaddr)
                 : "memory");
}

// ---------------------------------------------------------------------------
// Weight normalization: w[o,k] = (f16)( g[o] * v[o,k] / ||v[o,:]|| )
// One block (256 threads) per output row; padded rows (o >= Oreal) get zeros.
// ---------------------------------------------------------------------------
__global__ __launch_bounds__(256) void snn_wnorm_kernel(
    const float* __restrict__ v, const float* __restrict__ g,
    _Float16* __restrict__ w, int Oreal, int K)
{
    __shared__ float red[256];
    const int o   = blockIdx.x;
    const int tid = threadIdx.x;

    float s = 0.f;
    if (o < Oreal) {
        for (int k = tid; k < K; k += 256) {
            float x = v[(size_t)o * K + k];
            s += x * x;
        }
    }
    red[tid] = s;
    __syncthreads();
    for (int st = 128; st > 0; st >>= 1) {
        if (tid < st) red[tid] += red[tid + st];
        __syncthreads();
    }
    const float inv_norm = rsqrtf(red[0] + 1e-30f);
    const float scale = (o < Oreal) ? g[o] * inv_norm : 0.f;
    if (o < Oreal) {
        for (int k = tid; k < K; k += 256)
            w[(size_t)o * K + k] = (_Float16)(v[(size_t)o * K + k] * scale);
    } else {
        for (int k = tid; k < K; k += 256)
            w[(size_t)o * K + k] = (_Float16)0.f;
    }
}

// ---------------------------------------------------------------------------
// Input spikes: [B][K][T] f32  ->  [B][TP][K] f16 (zero fill for t in [T,TP))
// 32x32 LDS tile transpose, coalesced load (along t) and store (along k).
// ---------------------------------------------------------------------------
__global__ __launch_bounds__(256) void snn_transpose_convert_kernel(
    const float* __restrict__ in, _Float16* __restrict__ out,
    int K, int T, int TP)
{
    __shared__ _Float16 tile[32][33];
    const int b  = blockIdx.z;
    const int t0 = blockIdx.x * 32;
    const int k0 = blockIdx.y * 32;
    const int tx = threadIdx.x;   // 0..31
    const int ty = threadIdx.y;   // 0..7

    const float* inb = in + (size_t)b * K * T;
    _Float16*   outb = out + (size_t)b * TP * K;

    #pragma unroll
    for (int r = ty; r < 32; r += 8) {
        const int k = k0 + r;
        const int t = t0 + tx;
        float val = (t < T) ? inb[(size_t)k * T + t] : 0.f;
        tile[r][tx] = (_Float16)val;
    }
    __syncthreads();
    #pragma unroll
    for (int r = ty; r < 32; r += 8) {
        const int t = t0 + r;
        const int k = k0 + tx;
        if (t < TP) outb[(size_t)t * K + k] = tile[tx][r];
    }
}

// ---------------------------------------------------------------------------
// WMMA GEMM:  Z[b][m][n] = sum_k X[b][m][k] * W[n][k]
//   X : [B][MP][K]  f16  (activations, t-major)
//   W : [N][K]      f16  (row-major normalized weights -> B-operand K-major)
//   Z : [B][MP][N]  f32
// Block: 256 threads (8 waves), tile BM=128 x BN=64 x BK=32.
// Waves 4(M) x 2(N); each wave: 2x2 tiles of v_wmma_f32_16x16x32_f16.
// Double-buffered LDS staged with GLOBAL_LOAD_ASYNC_TO_LDS_B128 (ASYNCcnt):
// tile k+1 streams in while tile k runs on the matrix pipes.
// A-row indices are CLAMPED (not zero-filled): an A row only contributes to
// its own output row, and out-of-range output rows are masked at the store,
// so the main loop is branch-free and EXEC stays all-ones for WMMA.
// ---------------------------------------------------------------------------
__global__ __launch_bounds__(256) void snn_gemm_wmma_kernel(
    const _Float16* __restrict__ X, const _Float16* __restrict__ W,
    float* __restrict__ Z, int MP, int N, int K)
{
    __shared__ _Float16 sA[2][BM][BK];   // 2 x 8 KB  [t][k]
    __shared__ _Float16 sB[2][BN][BK];   // 2 x 4 KB  [o][k]

    const int b    = blockIdx.z;
    const int m0   = blockIdx.x * BM;
    const int n0   = blockIdx.y * BN;
    const int tid  = threadIdx.x;
    const int lane = tid & 31;
    const int wave = tid >> 5;          // 0..7
    const int wm   = wave >> 1;         // 0..3 : 32-row M strip
    const int wn   = wave & 1;          // 0..1 : 32-col N strip
    const int lhalf = lane >> 4;        // 0/1
    const int lrow  = lane & 15;        // 0..15

    const _Float16* Xb = X + (size_t)b * MP * K;
    float*          Zb = Z + (size_t)b * MP * N;

    // staging assignment: 16 bytes (8 f16) per thread per chunk
    const int ar = tid >> 2;            // 0..63 (A: two 64-row chunks)
    const int ac = (tid & 3) * 8;       // 0,8,16,24
    const int br = tid >> 2;            // 0..63 (B: 64 rows x 32 cols)
    const int bc = (tid & 3) * 8;

    // clamped source rows (branch-free; see header comment)
    const int gmA0 = (m0 + ar      < MP) ? (m0 + ar)      : (MP - 1);
    const int gmA1 = (m0 + ar + 64 < MP) ? (m0 + ar + 64) : (MP - 1);
    const _Float16* gA0 = Xb + (size_t)gmA0 * K + ac;
    const _Float16* gA1 = Xb + (size_t)gmA1 * K + ac;
    const _Float16* gB  = W  + (size_t)(n0 + br) * K + bc;

    v8f acc[2][2] = {{{}, {}}, {{}, {}}};

    const int nK = K / BK;

    // prologue: stage tile 0 into buffer 0
    async_copy_b128(&sA[0][ar][ac],      gA0);
    async_copy_b128(&sA[0][ar + 64][ac], gA1);
    async_copy_b128(&sB[0][br][bc],      gB);

    for (int kt = 0; kt < nK; ++kt) {
        const int cur = kt & 1;

        if (kt + 1 < nK) {
            // stage next tile into the other buffer, then wait only for the
            // 3 older (current-tile) async copies: in-order completion means
            // "<= 3 outstanding" == current tile resident in LDS.
            const int koff = (kt + 1) * BK;
            async_copy_b128(&sA[cur ^ 1][ar][ac],      gA0 + koff);
            async_copy_b128(&sA[cur ^ 1][ar + 64][ac], gA1 + koff);
            async_copy_b128(&sB[cur ^ 1][br][bc],      gB  + koff);
            asm volatile("s_wait_asynccnt 0x3" ::: "memory");
        } else {
            asm volatile("s_wait_asynccnt 0x0" ::: "memory");
        }
        __syncthreads();   // current tile visible to all waves

        // ---- load fragments per ISA VGPR layouts ----
        // A (16x32 f16): lane 0-15 row M=lane, K {0..7}|{16..23};
        //                lane 16-31 same rows, K {8..15}|{24..31}
        v16h afrag[2], bfrag[2];
        #pragma unroll
        for (int i = 0; i < 2; ++i) {
            const int row = wm * 32 + i * 16 + lrow;
            ((f32x4*)&afrag[i])[0] = *(const f32x4*)&sA[cur][row][lhalf * 8];
            ((f32x4*)&afrag[i])[1] = *(const f32x4*)&sA[cur][row][16 + lhalf * 8];
        }
        // B (32x16 f16): lane holds column N=lane%16, K {0..7}|{16..23} (lanes 0-15), {8..15}|{24..31} (lanes 16-31)
        #pragma unroll
        for (int j = 0; j < 2; ++j) {
            const int col = wn * 32 + j * 16 + lrow;
            ((f32x4*)&bfrag[j])[0] = *(const f32x4*)&sB[cur][col][lhalf * 8];
            ((f32x4*)&bfrag[j])[1] = *(const f32x4*)&sB[cur][col][16 + lhalf * 8];
        }

        #pragma unroll
        for (int i = 0; i < 2; ++i)
            #pragma unroll
            for (int j = 0; j < 2; ++j)
                acc[i][j] = __builtin_amdgcn_wmma_f32_16x16x32_f16(
                    false, afrag[i], false, bfrag[j],
                    (short)0, acc[i][j], false, false);

        __syncthreads();   // all reads of 'cur' done before it is re-staged
    }

    // ---- store C: VGPR r -> (M = r + 8*(lane/16), N = lane%16) ----
    #pragma unroll
    for (int i = 0; i < 2; ++i) {
        #pragma unroll
        for (int j = 0; j < 2; ++j) {
            const int col = n0 + wn * 32 + j * 16 + lrow;
            #pragma unroll
            for (int r = 0; r < 8; ++r) {
                const int row = m0 + wm * 32 + i * 16 + r + 8 * lhalf;
                if (row < MP) Zb[(size_t)row * N + col] = acc[i][j][r];
            }
        }
    }
}

// ---------------------------------------------------------------------------
// CUBA LIF scan over time. Z: [B][TP][C] f32 currents.
//   Xout (layers 1,2): [B][TP][C] f16 spikes (pad rows [T,TP) zeroed)
//   Yout (layer 3):    [B][Creal][T] f32 spikes (final output layout)
// Thread per (b, c); loop over t is coalesced across c.
// ---------------------------------------------------------------------------
__global__ void snn_lif_kernel(
    const float* __restrict__ Z, _Float16* __restrict__ Xout,
    float* __restrict__ Yout,
    const float* __restrict__ cd_arr, const float* __restrict__ vd_arr,
    int layer, int C, int Creal, int TP, int T, int B)
{
    const int idx = blockIdx.x * blockDim.x + threadIdx.x;
    const int c = idx % C;
    const int b = idx / C;
    if (b >= B) return;

    const float cd = cd_arr[layer];
    const float vd = vd_arr[layer];
    const float ka = 1.f - cd;
    const float kb = 1.f - vd;

    const float* z = Z + ((size_t)b * TP) * C + c;
    float cur = 0.f, volt = 0.f;
    for (int t = 0; t < T; ++t) {
        cur  = ka * cur + z[(size_t)t * C];
        volt = kb * volt + cur;
        const float s = (volt >= 1.25f) ? 1.f : 0.f;
        volt *= (1.f - s);
        if (Xout) Xout[((size_t)b * TP + t) * C + c] = (_Float16)s;
        if (Yout && c < Creal) Yout[((size_t)b * Creal + c) * T + t] = s;
    }
    if (Xout) {
        for (int t = T; t < TP; ++t)
            Xout[((size_t)b * TP + t) * C + c] = (_Float16)0.f;
    }
}

// ---------------------------------------------------------------------------
extern "C" void kernel_launch(void* const* d_in, const int* in_sizes, int n_in,
                              void* d_out, int out_size, void* d_ws, size_t ws_size,
                              hipStream_t stream)
{
    const float* spike = (const float*)d_in[0];
    const float* v1    = (const float*)d_in[1];
    const float* g1    = (const float*)d_in[2];
    const float* v2    = (const float*)d_in[3];
    const float* g2    = (const float*)d_in[4];
    const float* v3    = (const float*)d_in[5];
    const float* g3    = (const float*)d_in[6];
    const float* cds   = (const float*)d_in[7];
    const float* vds   = (const float*)d_in[8];

    constexpr int B = 16, K1 = 64 * 64 * 2, C = 512;
    constexpr int O3r = 100, O3p = 128;            // layer-3 N padded to 128
    constexpr int T = 300, TP = 304;               // time padded to 16

    // workspace carve-out (256B aligned)
    char* ws = (char*)d_ws;
    size_t off = 0;
    auto carve = [&](size_t bytes) {
        void* p = ws + off;
        off = (off + bytes + 255) & ~(size_t)255;
        return p;
    };
    _Float16* W1 = (_Float16*)carve((size_t)C   * K1 * 2);   //  8.0 MB
    _Float16* W2 = (_Float16*)carve((size_t)C   * C  * 2);   //  0.5 MB
    _Float16* W3 = (_Float16*)carve((size_t)O3p * C  * 2);   //  0.13 MB
    _Float16* X1 = (_Float16*)carve((size_t)B * TP * K1 * 2);//  79.7 MB
    _Float16* X2 = (_Float16*)carve((size_t)B * TP * C * 2); //  5.0 MB
    _Float16* X3 = (_Float16*)carve((size_t)B * TP * C * 2); //  5.0 MB
    float*    Zf = (float*)   carve((size_t)B * TP * C * 4); // 10.0 MB (reused)

    // --- normalized f16 weights ---
    snn_wnorm_kernel<<<C,   256, 0, stream>>>(v1, g1, W1, C,   K1);
    snn_wnorm_kernel<<<C,   256, 0, stream>>>(v2, g2, W2, C,   C);
    snn_wnorm_kernel<<<O3p, 256, 0, stream>>>(v3, g3, W3, O3r, C);

    // --- input spikes: [B][K1][T] f32 -> [B][TP][K1] f16 ---
    snn_transpose_convert_kernel<<<dim3((TP + 31) / 32, K1 / 32, B),
                                   dim3(32, 8), 0, stream>>>(spike, X1, K1, T, TP);

    const int gm = (TP + BM - 1) / BM;   // 3 M-blocks over t

    // --- layer 1: [TP x K1] @ [K1 x 512] ---
    snn_gemm_wmma_kernel<<<dim3(gm, C / BN, B), 256, 0, stream>>>(X1, W1, Zf, TP, C, K1);
    snn_lif_kernel<<<(B * C + 255) / 256, 256, 0, stream>>>(
        Zf, X2, nullptr, cds, vds, 0, C, C, TP, T, B);

    // --- layer 2: [TP x 512] @ [512 x 512] ---
    snn_gemm_wmma_kernel<<<dim3(gm, C / BN, B), 256, 0, stream>>>(X2, W2, Zf, TP, C, C);
    snn_lif_kernel<<<(B * C + 255) / 256, 256, 0, stream>>>(
        Zf, X3, nullptr, cds, vds, 1, C, C, TP, T, B);

    // --- layer 3: [TP x 512] @ [512 x 128(pad)] -> d_out [B][100][300] ---
    snn_gemm_wmma_kernel<<<dim3(gm, O3p / BN, B), 256, 0, stream>>>(X3, W3, Zf, TP, O3p, C);
    snn_lif_kernel<<<(B * O3p + 255) / 256, 256, 0, stream>>>(
        Zf, nullptr, (float*)d_out, cds, vds, 2, O3p, O3r, TP, T, B);
}
